// Decoder_55688545960079
// MI455X (gfx1250) — matrix operand order repaired and checked by reading, not compiled, not run on hardware
//
#include <hip/hip_runtime.h>
#include <hip/hip_bf16.h>

// ---------------------------------------------------------------------------
// CDNA5 (gfx1250) decoder step: additive attention + 4-layer GRU + FC.
// All matmuls use v_wmma_f32_16x16x32_bf16 (bf16 in, fp32 accum).
//  * Weights pre-converted to bf16 once per call.
//  * Energy kernel: 128-row LDS-blocked GEMM (Uw read once per 128 rows),
//    fused branch-free tanh/aw reduction.
//  * Generic GEMM: A tile staged once per block in LDS (bf16); waves share it
//    and each computes NT col-tiles -> k-loop is ds_load_b128 + wmma only.
// ---------------------------------------------------------------------------

typedef __bf16 bf16;
typedef __attribute__((ext_vector_type(16))) __bf16 v16bf;
typedef __attribute__((ext_vector_type(8)))  float  v8f;

union FragBF {
    v16bf v;
    bf16  e[16];
    uint4 q[2];
};

constexpr int S_   = 256;
constexpr int B_   = 512;
constexpr int H_   = 512;
constexpr int E_   = 256;
constexpr int OUT_ = 128;
constexpr int L_   = 4;
constexpr int G3H  = 3 * H_;    // 1536
constexpr int RIN  = H_ + E_;   // 768

static __device__ __forceinline__
v8f wmma_bf16(const FragBF& a, const FragBF& b, v8f c) {
    return __builtin_amdgcn_wmma_f32_16x16x32_bf16(
               false, a.v, false, b.v, (short)0, c, false, false);
}

// branch-free tanh: exp argument <= 0 so no overflow; v_exp + v_rcp, no EXEC
// divergence (libm tanhf expands to a branchy range-reduced path).
static __device__ __forceinline__ float fast_tanh(float x) {
    const float t = __expf(-2.0f * fabsf(x));
    const float r = (1.0f - t) * __builtin_amdgcn_rcpf(1.0f + t);
    return copysignf(r, x);
}

static __device__ __forceinline__ float fast_sigmoid(float x) {
    return __builtin_amdgcn_rcpf(1.0f + __expf(-x));
}

// ---------------------------------------------------------------------------
// fp32 -> bf16 bulk convert (n must be a multiple of 4)
// ---------------------------------------------------------------------------
__global__ __launch_bounds__(256)
void cvt_f32_bf16(const float* __restrict__ in, bf16* __restrict__ out, int n4)
{
    const int i = blockIdx.x * 256 + threadIdx.x;
    if (i < n4) {
        const float4 f = reinterpret_cast<const float4*>(in)[i];
        union { bf16 e[4]; uint2 u; } o;
        o.e[0] = (bf16)f.x; o.e[1] = (bf16)f.y; o.e[2] = (bf16)f.z; o.e[3] = (bf16)f.w;
        reinterpret_cast<uint2*>(out)[i] = o.u;
    }
}

// ---------------------------------------------------------------------------
// C[M,N] = A[M,K] * W[N,K]^T + bias[N].  A fp32 (staged to LDS bf16 once per
// block), W bf16 (pre-converted). 256 threads = 8 waves; each wave computes
// NT 16x16 tiles sharing one A fragment.
// Requires M%16==0, N%(128*NT)==0, K%32==0, K<=768.
// Fragment layouts used here (A/B k order is contiguous, same for both; not the ISA table order):
//   A (16x32): lane l -> row l&15; half h=l>>4: K = h*16+0..15
//   B (32x16): lane l -> col l&15; half h=l>>4: K = h*16+0..15
//   C (16x16): lane l, vgpr v -> row v + 8*(l>>4), col l&15
// ---------------------------------------------------------------------------
template<int NT>
__global__ __launch_bounds__(256)
void gemm_bias_wmma(const float* __restrict__ A, int lda,
                    const bf16* __restrict__ W, int ldw,
                    const float* __restrict__ bias,
                    float* __restrict__ C, int ldc, int K)
{
    __shared__ bf16 lA[16 * (768 + 8)];   // K<=768, pitch K+8 -> bank step 4
    const int pitch = K + 8;

    const int tid  = threadIdx.x;
    const int lane = tid & 31;
    const int wave = tid >> 5;
    const int r0   = blockIdx.x * 16;
    const int c0   = blockIdx.y * (128 * NT) + wave * (16 * NT);
    const int row  = lane & 15;
    const int kh   = lane >> 4;

    // stage A tile (16 x K) -> LDS bf16 (float4 loads, uint2 stores)
    const int Kq = K >> 2;
    for (int r = 0; r < 16; ++r) {
        for (int k4 = tid; k4 < Kq; k4 += 256) {
            const float4 f = *reinterpret_cast<const float4*>(
                                 A + (size_t)(r0 + r) * lda + k4 * 4);
            union { bf16 e[4]; uint2 u; } o;
            o.e[0] = (bf16)f.x; o.e[1] = (bf16)f.y;
            o.e[2] = (bf16)f.z; o.e[3] = (bf16)f.w;
            *reinterpret_cast<uint2*>(&lA[r * pitch + k4 * 4]) = o.u;
        }
    }
    __syncthreads();

    v8f acc[NT];
    #pragma unroll
    for (int t = 0; t < NT; ++t) acc[t] = (v8f){};

    const bf16* Abase = &lA[row * pitch];
    #pragma unroll 2
    for (int k0 = 0; k0 < K; k0 += 32) {
        FragBF af;
        af.q[0] = *reinterpret_cast<const uint4*>(Abase + k0 + kh * 16);
        af.q[1] = *reinterpret_cast<const uint4*>(Abase + k0 + kh * 16 + 8);
        #pragma unroll
        for (int t = 0; t < NT; ++t) {
            const bf16* Wr = W + (size_t)(c0 + t * 16 + row) * ldw + k0 + kh * 16;
            FragBF bf;
            bf.q[0] = *reinterpret_cast<const uint4*>(Wr);
            bf.q[1] = *reinterpret_cast<const uint4*>(Wr + 8);
            acc[t] = wmma_bf16(af, bf, acc[t]);
        }
    }

    #pragma unroll
    for (int t = 0; t < NT; ++t) {
        const int col = c0 + t * 16 + row;
        const float bb = bias ? bias[col] : 0.0f;
        #pragma unroll
        for (int v = 0; v < 8; ++v)
            C[(size_t)(r0 + v + 8 * kh) * ldc + col] = acc[t][v] + bb;
    }
}

// ---------------------------------------------------------------------------
// energy[s,b] = sum_o tanh( enc[s,b,:]·Uw[o,:] + Ub[o] + Wh[b,o] )*aw[o] + ab
// One workgroup per 128 consecutive (s,b) rows. Encoder tile (bf16) stays in
// LDS for the whole kernel; Uw is staged in 32-wide o-chunks (read ONCE per
// 128 rows). 8 waves; wave w owns m-tile w and both o-tiles of the chunk.
// ---------------------------------------------------------------------------
constexpr int PMA = 520;   // lA row pitch (bf16), 1040 B -> bank step 4
constexpr int PMB = 520;   // lB row pitch (bf16)
constexpr int PWH = 36;    // lWh row pitch (f32), 16B aligned

__global__ __launch_bounds__(256)
void attn_energy_wmma(const float* __restrict__ enc,   // [S,B,H]
                      const bf16*  __restrict__ Uwb,   // [H,H] bf16 [o,h]
                      const float* __restrict__ Ub,    // [H]
                      const float* __restrict__ Wh,    // [B,H]
                      const float* __restrict__ aw,    // [H]
                      const float* __restrict__ ab,    // [1]
                      float* __restrict__ energy)      // [S,B]
{
    __shared__ bf16  lA[128 * PMA];      // 133,120 B
    __shared__ bf16  lB[32 * PMB];       //  33,280 B
    __shared__ float lWh[128 * PWH];     //  18,432 B
    __shared__ float lAw[H_];
    __shared__ float lE[128];

    const int tid = threadIdx.x;
    const int r0  = blockIdx.x * 128;    // flattened (s,b) row
    const int s   = r0 >> 9;
    const int b0  = r0 & (B_ - 1);

    // encoder tile -> LDS bf16 (float4 loads, uint2 stores)
    for (int i = tid; i < 128 * (H_ / 4); i += 256) {
        const int r  = i >> 7;           // H_/4 == 128
        const int h4 = (i & 127) * 4;
        const float4 f = *reinterpret_cast<const float4*>(
                             enc + (size_t)(r0 + r) * H_ + h4);
        union { bf16 e[4]; uint2 u; } o;
        o.e[0] = (bf16)f.x; o.e[1] = (bf16)f.y; o.e[2] = (bf16)f.z; o.e[3] = (bf16)f.w;
        *reinterpret_cast<uint2*>(&lA[r * PMA + h4]) = o.u;
    }
    lAw[tid]       = aw[tid];
    lAw[tid + 256] = aw[tid + 256];
    if (tid < 128) lE[tid] = 0.0f;

    const int lane = tid & 31;
    const int wave = tid >> 5;
    const int row  = lane & 15;
    const int kh   = lane >> 4;

    float pe[8] = {0.f, 0.f, 0.f, 0.f, 0.f, 0.f, 0.f, 0.f};

    for (int oc = 0; oc < H_ / 32; ++oc) {
        __syncthreads();   // protect lB/lWh reuse (and initial lA fill)

        // stage Uw chunk: rows o = oc*32 .. +31 (bf16, uint4 copies)
        for (int i = tid; i < 32 * (H_ / 8); i += 256) {
            const int r = i >> 6;                // H_/8 == 64
            const int c = (i & 63) * 8;
            *reinterpret_cast<uint4*>(&lB[r * PMB + c]) =
                *reinterpret_cast<const uint4*>(&Uwb[(size_t)(oc * 32 + r) * H_ + c]);
        }
        // stage Wh + Ub chunk (128 rows x 32 cols)
        for (int i = tid; i < 128 * 8; i += 256) {
            const int r  = i >> 3;
            const int o4 = (i & 7) * 4;
            const float4 w = *reinterpret_cast<const float4*>(
                                 Wh + (size_t)(b0 + r) * H_ + oc * 32 + o4);
            const float4 u = *reinterpret_cast<const float4*>(Ub + oc * 32 + o4);
            float4 t;
            t.x = w.x + u.x; t.y = w.y + u.y; t.z = w.z + u.z; t.w = w.w + u.w;
            *reinterpret_cast<float4*>(&lWh[r * PWH + o4]) = t;
        }
        __syncthreads();

        v8f acc0 = {}, acc1 = {};
        const bf16* Abase = &lA[(wave * 16 + row) * PMA];
        const bf16* B0    = &lB[row * PMB];
        const bf16* B1    = &lB[(16 + row) * PMB];
        #pragma unroll 2
        for (int k0 = 0; k0 < H_; k0 += 32) {
            FragBF af, bf0, bf1;
            af.q[0]  = *reinterpret_cast<const uint4*>(Abase + k0 + kh * 16);
            af.q[1]  = *reinterpret_cast<const uint4*>(Abase + k0 + kh * 16 + 8);
            bf0.q[0] = *reinterpret_cast<const uint4*>(B0 + k0 + kh * 16);
            bf0.q[1] = *reinterpret_cast<const uint4*>(B0 + k0 + kh * 16 + 8);
            bf1.q[0] = *reinterpret_cast<const uint4*>(B1 + k0 + kh * 16);
            bf1.q[1] = *reinterpret_cast<const uint4*>(B1 + k0 + kh * 16 + 8);
            acc0 = wmma_bf16(af, bf0, acc0);
            acc1 = wmma_bf16(af, bf1, acc1);
        }

        // fused branch-free tanh / aw reduction for this o-chunk
        #pragma unroll
        for (int v = 0; v < 8; ++v) {
            const int rb = wave * 16 + v + 8 * kh;     // row within 128-tile
            pe[v] += fast_tanh(acc0[v] + lWh[rb * PWH + row])      * lAw[oc * 32 + row]
                   + fast_tanh(acc1[v] + lWh[rb * PWH + 16 + row]) * lAw[oc * 32 + 16 + row];
        }
    }

    #pragma unroll
    for (int v = 0; v < 8; ++v)
        atomicAdd(&lE[wave * 16 + v + 8 * kh], pe[v]);   // ds_add_f32
    __syncthreads();
    if (tid < 128)
        energy[s * B_ + b0 + tid] = lE[tid] + ab[0];
}

// softmax over the S (=256) dimension; one block per batch column.
__global__ __launch_bounds__(256)
void softmax_seq(float* __restrict__ e)   // [S,B] in place
{
    __shared__ float red[256];
    const int b = blockIdx.x;
    const int t = threadIdx.x;            // == s
    const float v = e[t * B_ + b];
    red[t] = v;
    __syncthreads();
    for (int off = 128; off > 0; off >>= 1) {
        if (t < off) red[t] = fmaxf(red[t], red[t + off]);
        __syncthreads();
    }
    const float mx = red[0];
    __syncthreads();
    const float p = __expf(v - mx);
    red[t] = p;
    __syncthreads();
    for (int off = 128; off > 0; off >>= 1) {
        if (t < off) red[t] += red[t + off];
        __syncthreads();
    }
    e[t * B_ + b] = p / red[0];
}

// context[b,h] = sum_s attn[s,b] * enc[s,b,h]  -> rnn_in[b, 0:H]
__global__ __launch_bounds__(256)
void context_kernel(const float* __restrict__ attn, const float* __restrict__ enc,
                    float* __restrict__ rnn_in)
{
    const int b = blockIdx.x >> 1;
    const int h = ((blockIdx.x & 1) << 8) + threadIdx.x;
    const float* ep = enc + (size_t)b * H_ + h;
    float acc = 0.f;
    #pragma unroll 4
    for (int s = 0; s < S_; ++s)
        acc += attn[s * B_ + b] * ep[(size_t)s * B_ * H_];
    rnn_in[b * RIN + h] = acc;
}

// rnn_in[b, H:H+E] = emb[x[b], :]
__global__ __launch_bounds__(256)
void embed_kernel(const int* __restrict__ x, const float* __restrict__ emb,
                  float* __restrict__ rnn_in)
{
    const int i = blockIdx.x * 256 + threadIdx.x;   // B*E
    const int b = i >> 8;
    const int e = i & (E_ - 1);
    rnn_in[b * RIN + H_ + e] = emb[x[b] * E_ + e];
}

// GRU gate math from precomputed gi = x@Wih^T+bih, gh = h@Whh^T+bhh
__global__ __launch_bounds__(256)
void gru_gate_kernel(const float* __restrict__ gi, const float* __restrict__ gh,
                     const float* __restrict__ hprev, float* __restrict__ hnew)
{
    const int i = blockIdx.x * 256 + threadIdx.x;   // B*H
    const int b = i >> 9;
    const int h = i & (H_ - 1);
    const float* gib = gi + (size_t)b * G3H;
    const float* ghb = gh + (size_t)b * G3H;
    const float r = fast_sigmoid(gib[h] + ghb[h]);
    const float z = fast_sigmoid(gib[H_ + h] + ghb[H_ + h]);
    const float n = fast_tanh(gib[2 * H_ + h] + r * ghb[2 * H_ + h]);
    hnew[i] = (1.f - z) * n + z * hprev[i];
}

// ---------------------------------------------------------------------------
extern "C" void kernel_launch(void* const* d_in, const int* in_sizes, int n_in,
                              void* d_out, int out_size, void* d_ws, size_t ws_size,
                              hipStream_t stream)
{
    (void)in_sizes; (void)n_in; (void)out_size; (void)ws_size;

    const int*   x      = (const int*)  d_in[0];
    const float* enc    = (const float*)d_in[1];
    const float* hidden = (const float*)d_in[2];   // [L,B,H]
    /* d_in[3] = cell, unused */
    const float* emb    = (const float*)d_in[4];
    const float* Uw     = (const float*)d_in[5];
    const float* Ub     = (const float*)d_in[6];
    const float* Ww     = (const float*)d_in[7];
    const float* Wb     = (const float*)d_in[8];
    const float* aw     = (const float*)d_in[9];
    const float* ab     = (const float*)d_in[10];
    const float* w_ih0  = (const float*)d_in[11];
    const float* w_hh0  = (const float*)d_in[12];
    const float* b_ih0  = (const float*)d_in[13];
    const float* b_hh0  = (const float*)d_in[14];
    const float* w_ih_r = (const float*)d_in[15];
    const float* w_hh_r = (const float*)d_in[16];
    const float* b_ih_r = (const float*)d_in[17];
    const float* b_hh_r = (const float*)d_in[18];
    const float* fcw    = (const float*)d_in[19];
    const float* fcb    = (const float*)d_in[20];

    float* out  = (float*)d_out;
    float* pred = out;                 // [B, OUT]
    float* hid  = out + B_ * OUT_;     // [L, B, H]

    // fp32 workspace
    float* ws     = (float*)d_ws;
    float* Wh     = ws;                   // B*H
    float* energy = Wh + B_ * H_;         // S*B
    float* rnn_in = energy + S_ * B_;     // B*RIN
    float* gi     = rnn_in + B_ * RIN;    // B*3H
    float* gh     = gi + B_ * G3H;        // B*3H
    // bf16 weight workspace (after fp32 region)
    bf16* Uwb   = (bf16*)(gh + B_ * G3H);
    bf16* Wwb   = Uwb   + H_ * H_;
    bf16* wih0b = Wwb   + H_ * H_;
    bf16* whh0b = wih0b + G3H * RIN;
    bf16* wihrb = whh0b + G3H * H_;
    bf16* whhrb = wihrb + (L_ - 1) * G3H * H_;
    bf16* fcwb  = whhrb + (L_ - 1) * G3H * H_;

    auto cvt = [&](const float* src, bf16* dst, int n) {
        cvt_f32_bf16<<<dim3((n / 4 + 255) / 256), 256, 0, stream>>>(src, dst, n / 4);
    };
    // 0. one-shot bf16 weight conversion (~39 MB traffic, ~2 us)
    cvt(Uw,     Uwb,   H_ * H_);
    cvt(Ww,     Wwb,   H_ * H_);
    cvt(w_ih0,  wih0b, G3H * RIN);
    cvt(w_hh0,  whh0b, G3H * H_);
    cvt(w_ih_r, wihrb, (L_ - 1) * G3H * H_);
    cvt(w_hh_r, whhrb, (L_ - 1) * G3H * H_);
    cvt(fcw,    fcwb,  OUT_ * H_);

    // 1. Wh = hidden[-1] @ Ww^T + Wb
    gemm_bias_wmma<4><<<dim3(B_ / 16, H_ / 512), 256, 0, stream>>>(
        hidden + 3 * B_ * H_, H_, Wwb, H_, Wb, Wh, H_, H_);

    // 2. energy (U never materialized; Uw read once per 128 rows)
    attn_energy_wmma<<<dim3(S_ * B_ / 128), 256, 0, stream>>>(
        enc, Uwb, Ub, Wh, aw, ab, energy);

    // 3. softmax over S (in place -> attention weights)
    softmax_seq<<<dim3(B_), 256, 0, stream>>>(energy);

    // 4. context -> rnn_in[:, 0:H]   5. embedding -> rnn_in[:, H:H+E]
    context_kernel<<<dim3(B_ * 2), 256, 0, stream>>>(energy, enc, rnn_in);
    embed_kernel<<<dim3(B_ * E_ / 256), 256, 0, stream>>>(x, emb, rnn_in);

    // 6. four GRU layers (single time step)
    for (int l = 0; l < L_; ++l) {
        const float* xin = (l == 0) ? rnn_in : (hid + (l - 1) * B_ * H_);
        const int    K   = (l == 0) ? RIN : H_;
        const bf16*  wih = (l == 0) ? wih0b : (wihrb + (size_t)(l - 1) * G3H * H_);
        const bf16*  whh = (l == 0) ? whh0b : (whhrb + (size_t)(l - 1) * G3H * H_);
        const float* bih = (l == 0) ? b_ih0 : (b_ih_r + (l - 1) * G3H);
        const float* bhh = (l == 0) ? b_hh0 : (b_hh_r + (l - 1) * G3H);

        gemm_bias_wmma<4><<<dim3(B_ / 16, G3H / 512), 256, 0, stream>>>(
            xin, K, wih, K, bih, gi, G3H, K);
        gemm_bias_wmma<4><<<dim3(B_ / 16, G3H / 512), 256, 0, stream>>>(
            hidden + l * B_ * H_, H_, whh, H_, bhh, gh, G3H, H_);
        gru_gate_kernel<<<dim3(B_ * H_ / 256), 256, 0, stream>>>(
            gi, gh, hidden + l * B_ * H_, hid + l * B_ * H_);
    }

    // 7. predictions = h3 @ fcw^T + fcb
    gemm_bias_wmma<1><<<dim3(B_ / 16, OUT_ / 128), 256, 0, stream>>>(
        hid + 3 * B_ * H_, H_, fcwb, H_, fcb, pred, OUT_, H_);
}
